// VideoMusicTransformer_V2_24489903522348
// MI455X (gfx1250) — matrix operand order repaired and checked by reading, not compiled, not run on hardware
//
#include <hip/hip_runtime.h>
#include <hip/hip_bf16.h>

// ---------------------------------------------------------------------------
// CDNA5 (gfx1250) implementation of the VideoMusicTransformer reference.
// All heavy math (KAN layers, attention, MoE experts) is lowered to
// v_wmma_f32_16x16x32_bf16 through one generic LDS-staged GEMM kernel.
// All operands are padded to K % 32 == 0 with zeroed pad columns so the
// GEMM inner loop is branch-free. The A-tile stage is software-pipelined
// one K-step ahead; if the toolchain exposes the gfx1250 async-to-LDS
// builtins we use GLOBAL_LOAD_ASYNC_TO_LDS_B64 + s_wait_asynccnt with
// double-buffered LDS, otherwise a register-staged pipeline.
// ---------------------------------------------------------------------------

typedef __bf16 bf16;
typedef __attribute__((ext_vector_type(16))) __bf16 v16bf;
typedef __attribute__((ext_vector_type(8)))  __bf16 v8bf;
typedef __attribute__((ext_vector_type(4)))  __bf16 v4bf;
typedef __attribute__((ext_vector_type(8)))  float  v8f;

#if __has_builtin(__builtin_amdgcn_global_load_async_to_lds_b64) && \
    __has_builtin(__builtin_amdgcn_s_wait_asynccnt)
#define ASYNC_LDS 1
// exact param type from clang diagnostic: int __attribute__((vector_size(8))) *
typedef int v2i __attribute__((vector_size(8)));
typedef __attribute__((address_space(1))) v2i* gv2i_p;
typedef __attribute__((address_space(3))) v2i* lv2i_p;
#else
#define ASYNC_LDS 0
#endif

#define NB      4
#define SEQ     300
#define NT      (NB*SEQ)       // 1200 tokens per stream
#define NTP     1216           // NT rounded up to 32 (A-row slack)
#define DMODEL  512
#define DFFN    1024
#define NH      8
#define DH      64
#define BHN     (NB*NH)        // 32 batched attention matrices
#define SKPAD   320            // padded key length (K for P*V, mult of 32)
#define VLD     320            // ld for V^T rows (mult of 32, zero padded)

// ---------------------------------------------------------------------------
// Generic bf16 WMMA GEMM:  C[M,N] (f32) = A[M,K] * Bw[N,K]^T (+bias)(+C)
// Block: 256 thr = 8 waves = (2 m-rows) x (4 n-groups); wave tile 16x64.
// Preconditions (guaranteed by producers):
//   Kpad % 32 == 0; lda >= Kpad, ldb >= Kpad (both % 8 == 0)
//   A readable for rows [0, ceil32(M)) with finite pad values;
//   A/B pad columns [Kreal, Kpad) are zero for valid rows.
// ---------------------------------------------------------------------------
__global__ __launch_bounds__(256) void k_gemm_bf16(
    const bf16* __restrict__ A, long long sA, int lda,
    const bf16* __restrict__ Bw, long long sB, int ldb,
    float* __restrict__ C, long long sC, int csplit, long long sC2, int ldc,
    int M, int N, int Kpad,
    const float* __restrict__ bias, int accum)
{
  const int bz = blockIdx.z;
  A  += (long long)bz * sA;
  Bw += (long long)bz * sB;
  if (csplit > 0) C += (long long)(bz / csplit) * sC + (long long)(bz % csplit) * sC2;
  else            C += (long long)bz * sC;

  const int tid  = threadIdx.x;
  const int wave = tid >> 5;
  const int lane = tid & 31;
  const int g    = lane >> 4;      // half-wave group
  const int ln   = lane & 15;
  const int mrow = wave >> 2;      // 0..1
  const int ngrp = wave & 3;       // 0..3

  const int mbase = blockIdx.y * 32;
  const int nbase = blockIdx.x * 256 + ngrp * 64;

  // A-stage source: 1 x 8B per thread per K-step
  const int srow = tid >> 3;            // 0..31
  const int skc  = (tid & 7) * 4;       // 0,4,..,28
  const bf16* astage = A + (long long)(mbase + srow) * lda + skc;

  // B row pointers (clamped once; ISA B layout: lo = k 8g.., hi = k 16+8g..)
  const bf16* bp[4];
  #pragma unroll
  for (int j = 0; j < 4; ++j) {
    int n = nbase + j * 16 + ln;
    if (n >= N) n = N - 1;
    bp[j] = Bw + (long long)n * ldb + 8 * g;
  }

  v8f acc[4];
  #pragma unroll
  for (int j = 0; j < 4; ++j)
    #pragma unroll
    for (int i = 0; i < 8; ++i) acc[j][i] = 0.0f;

#if ASYNC_LDS
  // ---- double-buffered LDS tiles fed by GLOBAL_LOAD_ASYNC_TO_LDS_B64 ----
  __shared__ __align__(16) bf16 As[2][32][40];
  // prologue: async-load tile for kk = 0 into buffer 0
  __builtin_amdgcn_global_load_async_to_lds_b64(
      (gv2i_p)(void*)astage, (lv2i_p)(void*)&As[0][srow][skc], 0, 0);
  __builtin_amdgcn_s_wait_asynccnt(0);
  __syncthreads();
  int p = 0;
  for (int kk = 0; kk < Kpad; kk += 32) {
    // issue next tile's async load into the other buffer (read-safe: all
    // waves passed the barrier after last consuming it)
    if (kk + 32 < Kpad)
      __builtin_amdgcn_global_load_async_to_lds_b64(
          (gv2i_p)(void*)(astage + kk + 32),
          (lv2i_p)(void*)&As[p ^ 1][srow][skc], 0, 0);

    const bf16* arow = &As[p][mrow * 16 + ln][0];
#else
  // ---- single LDS tile with register-staged pipeline ----
  __shared__ __align__(16) bf16 As[32][40];
  v4bf areg = *(const v4bf*)astage;       // preload kk = 0
  for (int kk = 0; kk < Kpad; kk += 32) {
    __syncthreads();                      // consumers of previous tile done
    *(v4bf*)&As[srow][skc] = areg;        // publish current tile
    if (kk + 32 < Kpad)                   // prefetch next tile into regs
      areg = *(const v4bf*)(astage + kk + 32);
    __syncthreads();

    const bf16* arow = &As[mrow * 16 + ln][0];
#endif

    // ---- A fragment from LDS (ISA A layout: lo = k 8g.., hi = k 16+8g..) ----
    v8bf alo = *(const v8bf*)(arow + 8 * g);
    v8bf ahi = *(const v8bf*)(arow + 16 + 8 * g);
    v16bf afrag;
    #pragma unroll
    for (int i = 0; i < 8; ++i) { afrag[i] = alo[i]; afrag[8 + i] = ahi[i]; }

    // ---- 4 B fragments + WMMA ----
    #pragma unroll
    for (int j = 0; j < 4; ++j) {
      const bf16* brow = bp[j] + kk;
      __builtin_prefetch(brow + 32, 0, 1);     // global_prefetch next K step
      v8bf blo = *(const v8bf*)(brow);
      v8bf bhi = *(const v8bf*)(brow + 16);
      v16bf bfrag;
      #pragma unroll
      for (int i = 0; i < 8; ++i) { bfrag[i] = blo[i]; bfrag[8 + i] = bhi[i]; }
      acc[j] = __builtin_amdgcn_wmma_f32_16x16x32_bf16(
          false, afrag, false, bfrag, (short)0, acc[j], false, false);
    }

#if ASYNC_LDS
    __builtin_amdgcn_s_wait_asynccnt(0);  // next tile landed (this wave)
    __syncthreads();                      // all waves: computed + landed
    p ^= 1;
  }
#else
  }
#endif

  // ---- epilogue: C/D layout m = r + 8g, n = lane%16 ----
  #pragma unroll
  for (int j = 0; j < 4; ++j) {
    int n = nbase + j * 16 + ln;
    if (n >= N) continue;
    float bv = bias ? bias[n] : 0.0f;
    #pragma unroll
    for (int r = 0; r < 8; ++r) {
      int m = mbase + mrow * 16 + g * 8 + r;
      if (m >= M) continue;
      float v = acc[j][r] + bv;
      long long ci = (long long)m * ldc + n;
      if (accum) v += C[ci];
      C[ci] = v;
    }
  }
}

// ---------------------------------------------------------------------------
// KAN feature expansion: F[m] = [ silu(x) (din) | bspline(x) (din*8) ]  (bf16)
// ---------------------------------------------------------------------------
__global__ void k_kan_expand(const float* __restrict__ x, bf16* __restrict__ F,
                             int M, int din, int ldF)
{
  long long i = (long long)blockIdx.x * blockDim.x + threadIdx.x;
  if (i >= (long long)M * din) return;
  int m = (int)(i / din), c = (int)(i % din);
  float t = x[(long long)m * din + c];
  bf16* fr = F + (long long)m * ldF;
  fr[c] = (bf16)(t / (1.0f + __expf(-t)));       // silu
  const float h = 0.4f;                           // 2/G
  float b[11];
  #pragma unroll
  for (int j = 0; j < 11; ++j) {
    float gl = (j - 3) * h - 1.0f;
    float gr = (j - 2) * h - 1.0f;
    b[j] = (t >= gl && t < gr) ? 1.0f : 0.0f;
  }
  #pragma unroll
  for (int o = 1; o <= 3; ++o) {
    float inv = 1.0f / (o * h);
    #pragma unroll
    for (int j = 0; j < 11 - o; ++j) {
      float gj   = (j - 3) * h - 1.0f;
      float gjo1 = (j - 3 + o + 1) * h - 1.0f;
      b[j] = (t - gj) * inv * b[j] + (gjo1 - t) * inv * b[j + 1];
    }
  }
  #pragma unroll
  for (int k = 0; k < 8; ++k) fr[din + c * 8 + k] = (bf16)b[k];
}

// zero pad columns [from, ld) of an M x ld bf16 matrix
__global__ void k_zero_cols(bf16* __restrict__ F, int M, int ld, int from)
{
  int pad = ld - from;
  long long i = (long long)blockIdx.x * blockDim.x + threadIdx.x;
  if (i >= (long long)M * pad) return;
  int m = (int)(i / pad), c = from + (int)(i % pad);
  F[(long long)m * ld + c] = (bf16)0.0f;
}

// Pack KAN weights: Wp[o] = [ base_w[o] (din) | spline_w[o] (din*8) ], pad 0.
__global__ void k_pack_kan(const float* __restrict__ base,
                           const float* __restrict__ spl,
                           bf16* __restrict__ W, int dout, int din, int ldw)
{
  long long i = (long long)blockIdx.x * blockDim.x + threadIdx.x;
  long long tot = (long long)dout * (din + 1);
  if (i >= tot) return;
  int o = (int)(i / (din + 1)), c = (int)(i % (din + 1));
  bf16* wr = W + (long long)o * ldw;
  if (c < din) {
    wr[c] = (bf16)base[(long long)o * din + c];
    const float* sp = spl + ((long long)o * din + c) * 8;
    #pragma unroll
    for (int k = 0; k < 8; ++k) wr[din + c * 8 + k] = (bf16)sp[k];
  } else {
    for (int k = 9 * din; k < ldw; ++k) wr[k] = (bf16)0.0f;  // zero pad cols
  }
}

__global__ void k_f2b(const float* __restrict__ x, bf16* __restrict__ y, long long n)
{
  long long i = (long long)blockIdx.x * blockDim.x + threadIdx.x;
  if (i < n) y[i] = (bf16)x[i];
}

// chord embedding + key concat: out[t] = emb_root[xr]+emb_attr[xa], col512=key
__global__ void k_embed_chord(const int* __restrict__ xroot, const int* __restrict__ xattr,
                              const float* __restrict__ er, const float* __restrict__ ea,
                              const float* __restrict__ fk, float* __restrict__ out)
{
  long long i = (long long)blockIdx.x * blockDim.x + threadIdx.x;
  if (i >= (long long)NT * 513) return;
  int t = (int)(i / 513), c = (int)(i % 513);
  float v;
  if (c < 512) v = er[xroot[t] * 512 + c] + ea[xattr[t] * 512 + c];
  else         v = fk[t / SEQ];
  out[i] = v;
}

__global__ void k_concat_vis(const float* __restrict__ fs, const float* __restrict__ so,
                             const float* __restrict__ fm, const float* __restrict__ fe,
                             float* __restrict__ out)
{
  long long i = (long long)blockIdx.x * blockDim.x + threadIdx.x;
  if (i >= (long long)NT * 776) return;
  int t = (int)(i / 776), c = (int)(i % 776);
  float v;
  if (c < 768)       v = fs[(long long)t * 768 + c];
  else if (c == 768) v = so[t];
  else if (c == 769) v = fm[t];
  else               v = fe[(long long)t * 6 + (c - 770)];
  out[i] = v;
}

// [NT,512] f32 -> [BH][SEQ][64] bf16 with optional RoPE
__global__ void k_qk_reshape(const float* __restrict__ X, bf16* __restrict__ out,
                             int use_rope)
{
  long long idx = (long long)blockIdx.x * blockDim.x + threadIdx.x;
  if (idx >= (long long)NT * NH * 32) return;
  int i = (int)(idx & 31);
  int h = (int)((idx >> 5) & 7);
  int t = (int)(idx >> 8);
  int b = t / SEQ, s = t % SEQ;
  const float* xr = X + (long long)t * DMODEL + h * DH;
  float x1 = xr[i], x2 = xr[i + 32];
  float o1, o2;
  if (use_rope) {
    float inv = __powf(10000.0f, -(2.0f * i) / 64.0f);
    float ang = s * inv;
    float cs = __cosf(ang), sn = __sinf(ang);
    o1 = x1 * cs - x2 * sn;
    o2 = x1 * sn + x2 * cs;
  } else { o1 = x1; o2 = x2; }
  bf16* orow = out + ((long long)(b * NH + h) * SEQ + s) * DH;
  orow[i] = (bf16)o1; orow[i + 32] = (bf16)o2;
}

// V [NT,512] f32 -> V^T bf16 [BH][64][VLD], zero-padded cols
__global__ void k_v_transpose(const float* __restrict__ V, bf16* __restrict__ Vt)
{
  long long idx = (long long)blockIdx.x * blockDim.x + threadIdx.x;
  if (idx >= (long long)BHN * DH * VLD) return;
  int s  = (int)(idx % VLD);
  int d  = (int)((idx / VLD) % DH);
  int bh = (int)(idx / ((long long)VLD * DH));
  int b = bh >> 3, h = bh & 7;
  float v = 0.0f;
  if (s < SEQ) v = V[((long long)(b * SEQ + s)) * DMODEL + h * DH + d];
  Vt[((long long)bh * DH + d) * VLD + s] = (bf16)v;
}

// scaled masked softmax over one score row -> bf16 probabilities (pads zeroed)
__global__ void k_softmax(const float* __restrict__ sc, bf16* __restrict__ P,
                          int causal)
{
  int row = blockIdx.x;                  // bh*SEQ + q
  int q = row % SEQ;
  const float* srow = sc + (long long)row * SKPAD;
  bf16* prow = P + (long long)row * SKPAD;
  const float scale = 0.125f;            // 1/sqrt(64)
  int tid = threadIdx.x, lane = tid & 31, wid = tid >> 5;
  __shared__ float red[4], red2[4];
  float mx = -3.4e38f;
  for (int k = tid; k < SEQ; k += 128) {
    float v = srow[k] * scale + ((causal && k > q) ? -1e9f : 0.0f);
    mx = fmaxf(mx, v);
  }
  for (int o = 16; o > 0; o >>= 1) mx = fmaxf(mx, __shfl_xor(mx, o));
  if (lane == 0) red[wid] = mx;
  __syncthreads();
  mx = fmaxf(fmaxf(red[0], red[1]), fmaxf(red[2], red[3]));
  float sum = 0.0f;
  for (int k = tid; k < SEQ; k += 128) {
    float v = srow[k] * scale + ((causal && k > q) ? -1e9f : 0.0f);
    sum += __expf(v - mx);
  }
  for (int o = 16; o > 0; o >>= 1) sum += __shfl_xor(sum, o);
  if (lane == 0) red2[wid] = sum;
  __syncthreads();
  sum = red2[0] + red2[1] + red2[2] + red2[3];
  float inv = 1.0f / sum;
  for (int k = tid; k < SKPAD; k += 128) {
    float pv = 0.0f;
    if (k < SEQ) {
      float v = srow[k] * scale + ((causal && k > q) ? -1e9f : 0.0f);
      pv = __expf(v - mx) * inv;
    }
    prow[k] = (bf16)pv;
  }
}

// layernorm of (x [+ r]) with affine g,b; optional bf16 mirror. D=512, 256 thr.
__global__ void k_layernorm(const float* __restrict__ x, const float* __restrict__ r,
                            const float* __restrict__ g, const float* __restrict__ bta,
                            float* __restrict__ out, bf16* __restrict__ outb)
{
  int row = blockIdx.x, tid = threadIdx.x;
  const float* xr = x + (long long)row * DMODEL;
  const float* rr = r ? r + (long long)row * DMODEL : nullptr;
  float v0 = xr[tid]       + (rr ? rr[tid]       : 0.0f);
  float v1 = xr[tid + 256] + (rr ? rr[tid + 256] : 0.0f);
  int lane = tid & 31, wid = tid >> 5;
  __shared__ float red[8];
  float s = v0 + v1;
  for (int o = 16; o > 0; o >>= 1) s += __shfl_xor(s, o);
  if (lane == 0) red[wid] = s;
  __syncthreads();
  float tot = 0.0f;
  #pragma unroll
  for (int i = 0; i < 8; ++i) tot += red[i];
  float mu = tot / DMODEL;
  float d0 = v0 - mu, d1 = v1 - mu;
  float vs = d0 * d0 + d1 * d1;
  for (int o = 16; o > 0; o >>= 1) vs += __shfl_xor(vs, o);
  __syncthreads();
  if (lane == 0) red[wid] = vs;
  __syncthreads();
  float var = 0.0f;
  #pragma unroll
  for (int i = 0; i < 8; ++i) var += red[i];
  var /= DMODEL;
  float inv = rsqrtf(var + 1e-5f);
  float o0 = g[tid] * d0 * inv + bta[tid];
  float o1 = g[tid + 256] * d1 * inv + bta[tid + 256];
  long long base = (long long)row * DMODEL;
  out[base + tid] = o0; out[base + tid + 256] = o1;
  if (outb) { outb[base + tid] = (bf16)o0; outb[base + tid + 256] = (bf16)o1; }
}

// gate logits: one wave per token, 6 dot products of length 512
__global__ void k_gate6(const float* __restrict__ x, const float* __restrict__ gw,
                        float* __restrict__ logits)
{
  int t = blockIdx.x, lane = threadIdx.x;
  const float* xr = x + (long long)t * DMODEL;
  for (int e = 0; e < 6; ++e) {
    float s = 0.0f;
    for (int c = lane; c < DMODEL; c += 32) s += xr[c] * gw[e * DMODEL + c];
    for (int o = 16; o > 0; o >>= 1) s += __shfl_xor(s, o);
    if (lane == 0) logits[t * 6 + e] = s;
  }
}

__global__ void k_top2(const float* __restrict__ logits, float* __restrict__ gates,
                       int* __restrict__ idxo)
{
  int t = blockIdx.x * blockDim.x + threadIdx.x;
  if (t >= NT) return;
  const float* l = logits + t * 6;
  int i1 = 0; float v1 = l[0];
  for (int e = 1; e < 6; ++e) if (l[e] > v1) { v1 = l[e]; i1 = e; }
  int i2 = 0; float v2 = -3.4e38f;
  for (int e = 0; e < 6; ++e) { if (e == i1) continue; if (l[e] > v2) { v2 = l[e]; i2 = e; } }
  float e2 = __expf(v2 - v1);
  gates[t * 2]     = 1.0f / (1.0f + e2);
  gates[t * 2 + 1] = e2 / (1.0f + e2);
  idxo[t * 2] = i1; idxo[t * 2 + 1] = i2;
}

// acc (+)= w(t) * eo ; e<0 => shared expert, weight 1
__global__ void k_moe_combine(const float* __restrict__ eo, float* __restrict__ acc,
                              const float* __restrict__ gates, const int* __restrict__ idx,
                              int e, int init)
{
  long long i = (long long)blockIdx.x * blockDim.x + threadIdx.x;
  if (i >= (long long)NT * DMODEL) return;
  int t = (int)(i >> 9);
  float w;
  if (e < 0) w = 1.0f;
  else {
    w = 0.0f;
    if (idx[t * 2] == e)     w += gates[t * 2];
    if (idx[t * 2 + 1] == e) w += gates[t * 2 + 1];
  }
  float v = w * eo[i];
  acc[i] = init ? v : acc[i] + v;
}

// ---------------------------------------------------------------------------
// Host orchestration
// ---------------------------------------------------------------------------
extern "C" void kernel_launch(void* const* d_in, const int* in_sizes, int n_in,
                              void* d_out, int out_size, void* d_ws, size_t ws_size,
                              hipStream_t stream)
{
  (void)in_sizes; (void)n_in; (void)out_size; (void)ws_size;
  hipStream_t st = stream;

  // --- data inputs (setup_inputs dict order) ---
  const int*   xroot = (const int*)d_in[1];
  const int*   xattr = (const int*)d_in[2];
  const float* fs    = (const float*)d_in[3];
  const float* fk    = (const float*)d_in[4];
  const float* so    = (const float*)d_in[5];
  const float* fm    = (const float*)d_in[6];
  const float* fe    = (const float*)d_in[7];

  // --- param leaves: jax tree_flatten (sorted dict keys). Index map:
  // dec layer L @ 31L: ca{bk,bo,bq,bv,wk,wo,wq,wv}=+0..7, ln1{b,g}=+8,9,
  //   ln2=+10,11, ln3=+12,13, moe{e.w1.b,e.w1.s,e.w2.b,e.w2.s,gate,
  //   sh.w1.b,sh.w1.s,sh.w2.b,sh.w2.s}=+14..22, sa{...}=+23..30
  // 124,125 dec_norm{b,g}; 126 emb_attr; 127 emb_root
  // enc layer L @ 128+21L: attn=+0..7, ln1=+8,9, ln2=+10,11, moe=+12..20
  // 212,213 enc_norm{b,g}; 214,215 lin_chord; 216,217 lin_vis; 218,219 wout
  auto Lp = [&](int leaf) -> const float* { return (const float*)d_in[8 + leaf]; };

  // --- workspace bump allocator ---
  char* wsB = (char*)d_ws; size_t off = 0;
  auto alloc = [&](size_t bytes) -> void* {
    void* p = wsB + off;
    off += bytes; off = (off + 255) & ~(size_t)255;
    return p;
  };
  float* catbuf = (float*)alloc((size_t)NT * 776 * 4);
  bf16*  FA     = (bf16*) alloc((size_t)NTP * 7008 * 2);  // expand(x): ld<=7008
  bf16*  FB     = (bf16*) alloc((size_t)NTP * 9216 * 2);  // expand(h): ld 9216
  bf16*  Wp     = (bf16*) alloc((size_t)1024 * 4640 * 2); // packed KAN weights
  bf16*  wb     = (bf16*) alloc((size_t)512 * 512 * 2);   // attn weight bf16
  float* memf   = (float*)alloc((size_t)NT * DMODEL * 4);
  float* ybuf   = (float*)alloc((size_t)NT * DMODEL * 4);
  bf16*  memBf  = (bf16*) alloc((size_t)NTP * DMODEL * 2);
  bf16*  inB    = (bf16*) alloc((size_t)NTP * DMODEL * 2);
  bf16*  aob    = (bf16*) alloc((size_t)NTP * DMODEL * 2);
  float* Qf     = (float*)alloc((size_t)NT * DMODEL * 4);
  float* Kf     = (float*)alloc((size_t)NT * DMODEL * 4);
  float* Vf     = (float*)alloc((size_t)NT * DMODEL * 4);
  bf16*  Qh     = (bf16*) alloc((size_t)(BHN * SEQ + 32) * DH * 2);   // +32 row slack
  bf16*  Kh     = (bf16*) alloc((size_t)BHN * SEQ * DH * 2);
  bf16*  Vt     = (bf16*) alloc((size_t)BHN * DH * VLD * 2);
  float* scb    = (float*)alloc((size_t)BHN * SEQ * SKPAD * 4);
  bf16*  Pb     = (bf16*) alloc((size_t)(BHN * SEQ + 32) * SKPAD * 2); // +32 row slack
  float* attnO  = (float*)alloc((size_t)NT * DMODEL * 4);
  float* tmp    = (float*)alloc((size_t)NT * DMODEL * 4);
  float* hbuf   = (float*)alloc((size_t)NT * DFFN * 4);
  float* eo     = (float*)alloc((size_t)NT * DMODEL * 4);
  float* macc   = (float*)alloc((size_t)NT * DMODEL * 4);
  float* logits = (float*)alloc((size_t)NT * 6 * 4);
  float* gates  = (float*)alloc((size_t)NT * 2 * 4);
  int*   tidx   = (int*)  alloc((size_t)NT * 2 * 4);

  auto gemm = [&](const bf16* A, long long sA, int lda,
                  const bf16* Bw, long long sB, int ldb,
                  float* C, long long sC, int csplit, long long sC2, int ldc,
                  int M, int N, int Kpad,
                  const float* bias, int accum, int batches) {
    dim3 g((N + 255) / 256, (M + 31) / 32, batches);
    k_gemm_bf16<<<g, 256, 0, st>>>(A, sA, lda, Bw, sB, ldb,
                                   C, sC, csplit, sC2, ldc,
                                   M, N, Kpad, bias, accum);
  };
  auto conv = [&](const float* src, bf16* dst, long long n) {
    k_f2b<<<(unsigned)((n + 255) / 256), 256, 0, st>>>(src, dst, n);
  };
  // expand + pack + GEMM = one KAN layer
  auto kan = [&](const float* x, int din, int dout,
                 const float* base, const float* spl, bf16* Fbuf, float* out) {
    int K = 9 * din, Kpad = (K + 31) & ~31;
    long long nE = (long long)NT * din;
    k_kan_expand<<<(unsigned)((nE + 255) / 256), 256, 0, st>>>(x, Fbuf, NT, din, Kpad);
    if (Kpad > K) {
      long long nZ = (long long)NT * (Kpad - K);
      k_zero_cols<<<(unsigned)((nZ + 255) / 256), 256, 0, st>>>(Fbuf, NT, Kpad, K);
    }
    long long nP = (long long)dout * (din + 1);
    k_pack_kan<<<(unsigned)((nP + 255) / 256), 256, 0, st>>>(base, spl, Wp, dout, din, Kpad);
    gemm(Fbuf, 0, Kpad, Wp, 0, Kpad, out, 0, 0, 0, dout,
         NT, dout, Kpad, nullptr, 0, 1);
  };

  struct AttnW { const float *bk,*bo,*bq,*bv,*wk,*wo,*wq,*wv; };
  auto attw = [&](int base) {
    AttnW w; w.bk=Lp(base+0); w.bo=Lp(base+1); w.bq=Lp(base+2); w.bv=Lp(base+3);
    w.wk=Lp(base+4); w.wo=Lp(base+5); w.wq=Lp(base+6); w.wv=Lp(base+7); return w;
  };

  auto mha = [&](const bf16* qB, const bf16* kvB, AttnW w, int causal, int rope,
                 float* outF) {
    const long long WN = (long long)512 * 512;
    // projections (token GEMMs, bias fused)
    conv(w.wq, wb, WN);
    gemm(qB, 0, DMODEL, wb, 0, DMODEL, Qf, 0, 0, 0, DMODEL,
         NT, DMODEL, DMODEL, w.bq, 0, 1);
    conv(w.wk, wb, WN);
    gemm(kvB, 0, DMODEL, wb, 0, DMODEL, Kf, 0, 0, 0, DMODEL,
         NT, DMODEL, DMODEL, w.bk, 0, 1);
    conv(w.wv, wb, WN);
    gemm(kvB, 0, DMODEL, wb, 0, DMODEL, Vf, 0, 0, 0, DMODEL,
         NT, DMODEL, DMODEL, w.bv, 0, 1);
    // head-major reshapes (+ RoPE)
    long long nQ = (long long)NT * NH * 32;
    k_qk_reshape<<<(unsigned)((nQ + 255) / 256), 256, 0, st>>>(Qf, Qh, rope);
    k_qk_reshape<<<(unsigned)((nQ + 255) / 256), 256, 0, st>>>(Kf, Kh, rope);
    long long nV = (long long)BHN * DH * VLD;
    k_v_transpose<<<(unsigned)((nV + 255) / 256), 256, 0, st>>>(Vf, Vt);
    // scores = Q K^T (batched over 32 (b,h))
    gemm(Qh, (long long)SEQ * DH, DH, Kh, (long long)SEQ * DH, DH,
         scb, (long long)SEQ * SKPAD, 0, 0, SKPAD,
         SEQ, SEQ, DH, nullptr, 0, BHN);
    k_softmax<<<BHN * SEQ, 128, 0, st>>>(scb, Pb, causal);
    // out = P V  (C scattered back to [token, h*64+d])
    gemm(Pb, (long long)SEQ * SKPAD, SKPAD, Vt, (long long)DH * VLD, VLD,
         attnO, (long long)SEQ * DMODEL, NH, DH, DMODEL,
         SEQ, DH, SKPAD, nullptr, 0, BHN);
    // output projection
    conv(attnO, aob, (long long)NT * DMODEL);
    conv(w.wo, wb, WN);
    gemm(aob, 0, DMODEL, wb, 0, DMODEL, outF, 0, 0, 0, DMODEL,
         NT, DMODEL, DMODEL, w.bo, 0, 1);
  };

  auto moe = [&](const float* x, int mbase) {
    const float *ew1b = Lp(mbase+0), *ew1s = Lp(mbase+1);
    const float *ew2b = Lp(mbase+2), *ew2s = Lp(mbase+3);
    const float *gw   = Lp(mbase+4);
    const float *sw1b = Lp(mbase+5), *sw1s = Lp(mbase+6);
    const float *sw2b = Lp(mbase+7), *sw2s = Lp(mbase+8);
    k_gate6<<<NT, 32, 0, st>>>(x, gw, logits);
    k_top2<<<(NT + 255) / 256, 256, 0, st>>>(logits, gates, tidx);
    // expand(x) once, reuse across all 7 experts
    const int K1 = 9 * DMODEL;   // 4608 (already mult of 32)
    long long nE = (long long)NT * DMODEL;
    k_kan_expand<<<(unsigned)((nE + 255) / 256), 256, 0, st>>>(x, FA, NT, DMODEL, K1);
    long long nCmb = (long long)NT * DMODEL;
    for (int e = -1; e < 6; ++e) {     // -1 = shared expert
      const float* w1b = (e < 0) ? sw1b : ew1b + (long long)e * DFFN * DMODEL;
      const float* w1s = (e < 0) ? sw1s : ew1s + (long long)e * DFFN * DMODEL * 8;
      const float* w2b = (e < 0) ? sw2b : ew2b + (long long)e * DMODEL * DFFN;
      const float* w2s = (e < 0) ? sw2s : ew2s + (long long)e * DMODEL * DFFN * 8;
      long long nP = (long long)DFFN * (DMODEL + 1);
      k_pack_kan<<<(unsigned)((nP + 255) / 256), 256, 0, st>>>(w1b, w1s, Wp, DFFN, DMODEL, K1);
      gemm(FA, 0, K1, Wp, 0, K1, hbuf, 0, 0, 0, DFFN,
           NT, DFFN, K1, nullptr, 0, 1);
      kan(hbuf, DFFN, DMODEL, w2b, w2s, FB, eo);
      k_moe_combine<<<(unsigned)((nCmb + 255) / 256), 256, 0, st>>>(
          eo, macc, gates, tidx, e, e < 0 ? 1 : 0);
    }
  };

  // ---------------- front end ----------------
  {
    long long n = (long long)NT * 513;
    k_embed_chord<<<(unsigned)((n + 255) / 256), 256, 0, st>>>(
        xroot, xattr, Lp(127), Lp(126), fk, catbuf);
    kan(catbuf, 513, DMODEL, Lp(214), Lp(215), FA, ybuf);     // lin_chord -> y
    n = (long long)NT * 776;
    k_concat_vis<<<(unsigned)((n + 255) / 256), 256, 0, st>>>(fs, so, fm, fe, catbuf);
    kan(catbuf, 776, DMODEL, Lp(216), Lp(217), FA, memf);     // lin_vis -> mem
  }

  // ---------------- encoder ----------------
  for (int L = 0; L < 4; ++L) {
    int base = 128 + 21 * L;
    conv(memf, memBf, (long long)NT * DMODEL);
    mha(memBf, memBf, attw(base), /*causal=*/0, /*rope=*/1, tmp);
    k_layernorm<<<NT, 256, 0, st>>>(memf, tmp, Lp(base + 9), Lp(base + 8), memf, nullptr);
    moe(memf, base + 12);
    k_layernorm<<<NT, 256, 0, st>>>(memf, macc, Lp(base + 11), Lp(base + 10), memf, nullptr);
  }
  k_layernorm<<<NT, 256, 0, st>>>(memf, nullptr, Lp(213), Lp(212), memf, memBf);

  // ---------------- decoder ----------------
  for (int L = 0; L < 4; ++L) {
    int base = 31 * L;
    conv(ybuf, inB, (long long)NT * DMODEL);
    mha(inB, inB, attw(base + 23), /*causal=*/1, /*rope=*/1, tmp);   // self-attn
    k_layernorm<<<NT, 256, 0, st>>>(ybuf, tmp, Lp(base + 9), Lp(base + 8), ybuf, nullptr);
    conv(ybuf, inB, (long long)NT * DMODEL);
    mha(inB, memBf, attw(base + 0), /*causal=*/0, /*rope=*/0, tmp);  // cross-attn
    k_layernorm<<<NT, 256, 0, st>>>(ybuf, tmp, Lp(base + 11), Lp(base + 10), ybuf, nullptr);
    moe(ybuf, base + 14);
    k_layernorm<<<NT, 256, 0, st>>>(ybuf, macc, Lp(base + 13), Lp(base + 12), ybuf, nullptr);
  }
  k_layernorm<<<NT, 256, 0, st>>>(ybuf, nullptr, Lp(125), Lp(124), ybuf, nullptr);

  // ---------------- output head (KAN, 512 -> 160) ----------------
  kan(ybuf, DMODEL, 160, Lp(218), Lp(219), FA, (float*)d_out);
}
